// InductionNetwork_893353197971
// MI455X (gfx1250) — matrix operand order repaired and checked by reading, not compiled, not run on hardware
//
#include <hip/hip_runtime.h>
#include <hip/hip_bf16.h>

// ---------------------------------------------------------------------------
// Problem constants (from reference setup_inputs): B=256, C=20, K=5, Q=15,
// D=768, H=100.  M (d_in[4]) is unused by the reference computation.
// ---------------------------------------------------------------------------
#define BB   256
#define CC   20
#define KK   5
#define QQ   15
#define DD   768
#define HH   100
#define SROWS (BB * CC * KK)     // 25600 support rows
#define QROWS (BB * CC * QQ)     // 76800 query rows
#define QPB   (CC * QQ)          // 300 query rows per batch
#define RTILES ((QPB + 15) / 16) // 19 row tiles per batch

#define MT    4                  // m-tiles per GEMM1 block
#define MROWS (MT * 16)          // 64 support rows per block
#define NT    3                  // n-tiles per wave (16 waves * 3 = 48 = 768/16)

typedef __attribute__((ext_vector_type(16))) __bf16          v16bf;
typedef __attribute__((ext_vector_type(8)))  float           v8f;
typedef __attribute__((ext_vector_type(8)))  unsigned short  v8us;
typedef __attribute__((ext_vector_type(16))) unsigned short  v16us;

__device__ __forceinline__ unsigned short f2bf(float f) {
  // round-to-nearest-even f32 -> bf16
  unsigned int u = __float_as_uint(f);
  u += 0x7FFFu + ((u >> 16) & 1u);
  return (unsigned short)(u >> 16);
}

__device__ __forceinline__ v16bf mkbf16(v8us lo, v8us hi) {
  v16us u = __builtin_shufflevector(lo, hi, 0, 1, 2, 3, 4, 5, 6, 7,
                                    8, 9, 10, 11, 12, 13, 14, 15);
  return __builtin_bit_cast(v16bf, u);
}

// ---------------------------------------------------------------------------
// Kernel 0: W_w (D x D f32, e = S@W so W[k][n]) -> WT bf16 with WT[n][k].
// ---------------------------------------------------------------------------
__global__ void k_wt(const float* __restrict__ W, unsigned short* __restrict__ WT) {
  int idx = blockIdx.x * 256 + threadIdx.x;
  if (idx >= DD * DD) return;
  int n = idx / DD, k = idx - n * DD;
  WT[idx] = f2bf(W[k * DD + n]);
}

// ---------------------------------------------------------------------------
// Kernel 1: e = support @ W + W_b, then squash per row -> e_hat (f32).
// Block = 512 threads (16 waves); block owns 64 support rows x all 768 cols.
// A tile (64x768 bf16 = 96 KB) lives in CDNA5's 320 KB LDS.  Each wave owns
// 3 n-tiles x 4 m-tiles; per K-step one B tile feeds 4 WMMAs (4x less L2
// traffic on W than a 16-row blocking: 472 MB total, under the HBM floor).
// ---------------------------------------------------------------------------
__global__ __launch_bounds__(512) void k_gemm1(const float* __restrict__ S,
                                               const unsigned short* __restrict__ WT,
                                               const float* __restrict__ Wb,
                                               float* __restrict__ ehat) {
  __shared__ __align__(16) unsigned short As[MROWS * DD]; // 96 KB bf16 A tile
  __shared__ float ssq[MROWS];
  __shared__ float sscale[MROWS];

  const int tid = threadIdx.x;
  const int m0  = blockIdx.x * MROWS;

  // Stage A tile: f32 -> bf16 into LDS (coalesced float4 reads, b64 LDS stores)
  {
    const float4* src = (const float4*)(S + (size_t)m0 * DD);
    for (int i = tid; i < MROWS * DD / 4; i += 512) {
      float4 v = src[i];
      unsigned long long p = (unsigned long long)f2bf(v.x)
                           | ((unsigned long long)f2bf(v.y) << 16)
                           | ((unsigned long long)f2bf(v.z) << 32)
                           | ((unsigned long long)f2bf(v.w) << 48);
      *(unsigned long long*)(&As[i * 4]) = p;
    }
    if (tid < MROWS) ssq[tid] = 0.0f;
  }
  __syncthreads();

  const int w  = tid >> 5;        // wave id 0..15
  const int l  = tid & 31;        // lane (wave32)
  const int lm = l & 15;
  const int lh = l >> 4;

  const v8f zero = {0.f, 0.f, 0.f, 0.f, 0.f, 0.f, 0.f, 0.f};
  v8f acc[NT][MT];
  for (int n = 0; n < NT; ++n)
    for (int m = 0; m < MT; ++m) acc[n][m] = zero;

  const unsigned short* brow[NT];
  for (int n = 0; n < NT; ++n)
    brow[n] = WT + (size_t)((w * NT + n) * 16 + lm) * DD;

  for (int kk = 0; kk < DD; kk += 32) {
    // A 16x32 bf16: lane half lh holds contiguous K chunk [kk+16lh, +16)
    v16bf a[MT];
    for (int m = 0; m < MT; ++m) {
      const unsigned short* ar = &As[(m * 16 + lm) * DD + kk];
      a[m] = mkbf16(*(const v8us*)(ar + lh * 16),
                    *(const v8us*)(ar + lh * 16 + 8));
    }
    for (int n = 0; n < NT; ++n) {
      // B 32x16 bf16: lane half lh holds contiguous K chunk [kk+16lh, +16)
      v16bf b = __builtin_bit_cast(v16bf, *(const v16us*)(brow[n] + kk + lh * 16));
      __builtin_prefetch(brow[n] + kk + 128, 0, 3);
      for (int m = 0; m < MT; ++m)
        acc[n][m] = __builtin_amdgcn_wmma_f32_16x16x32_bf16(
            false, a[m], false, b, (short)0, acc[n][m], false, false);
    }
  }

  // bias: column (w*NT+n)*16+lm for all rows of this lane
  for (int n = 0; n < NT; ++n) {
    float bias = Wb[(w * NT + n) * 16 + lm];
    for (int m = 0; m < MT; ++m)
      for (int v = 0; v < 8; ++v) acc[n][m][v] += bias;
  }

  // Row sum-of-squares: lane-local over NT tiles, shfl within 16-lane half
  // (C/D layout: lanes 0-15 hold M=v, lanes 16-31 hold M=v+8), then one
  // ds_add_f32 per row per wave.
  for (int m = 0; m < MT; ++m) {
    float part[8];
    for (int v = 0; v < 8; ++v) {
      float s = 0.f;
      for (int n = 0; n < NT; ++n) s += acc[n][m][v] * acc[n][m][v];
      part[v] = s;
    }
    for (int off = 1; off < 16; off <<= 1)
      for (int v = 0; v < 8; ++v) part[v] += __shfl_xor(part[v], off);
    if (lm == 0)
      for (int v = 0; v < 8; ++v)
        atomicAdd(&ssq[m * 16 + lh * 8 + v], part[v]);
  }
  __syncthreads();

  if (tid < MROWS) {
    float n = ssq[tid];
    sscale[tid] = n / ((1.0f + n) * sqrtf(1e-9f + n));
  }
  __syncthreads();

  for (int n = 0; n < NT; ++n) {
    const int col = (w * NT + n) * 16 + lm;
    for (int m = 0; m < MT; ++m)
      for (int v = 0; v < 8; ++v) {
        int row = m * 16 + lh * 8 + v;
        ehat[(size_t)(m0 + row) * DD + col] = acc[n][m][v] * sscale[row];
      }
  }
}

// ---------------------------------------------------------------------------
// Kernel 2: dynamic routing, one block per (b,c) pair.  3 iterations of
// softmax(5) -> c = sum_k di*eh -> squash -> b += eh . c.  Emits c_i as bf16.
// ---------------------------------------------------------------------------
__global__ __launch_bounds__(256) void k_route(const float* __restrict__ ehat,
                                               unsigned short* __restrict__ cib) {
  __shared__ float eh[KK * DD];  // 15 KB
  __shared__ float red[256];
  __shared__ float blog[KK];
  __shared__ float bacc[KK];

  const int tid = threadIdx.x;
  const size_t base = (size_t)blockIdx.x * KK * DD;
  for (int i = tid; i < KK * DD; i += 256) eh[i] = ehat[base + i];
  if (tid < KK) blog[tid] = 0.0f;
  __syncthreads();

  float cl[3]; // this thread's 3 feature slots of c_hat (d = tid + 256j)
  for (int it = 0; it < 3; ++it) {
    // softmax over K=5 (redundant per thread, values in LDS)
    float bl[KK], mx = -3.0e38f;
    for (int k = 0; k < KK; ++k) { bl[k] = blog[k]; mx = fmaxf(mx, bl[k]); }
    float di[KK], den = 0.f;
    for (int k = 0; k < KK; ++k) { di[k] = __expf(bl[k] - mx); den += di[k]; }
    float rden = 1.0f / den;
    for (int k = 0; k < KK; ++k) di[k] *= rden;

    // c[d] = sum_k di[k] * eh[k][d]
    float loc = 0.f;
    for (int j = 0; j < 3; ++j) {
      int d = tid + j * 256;
      float s = 0.f;
      for (int k = 0; k < KK; ++k) s += di[k] * eh[k * DD + d];
      cl[j] = s;
      loc += s * s;
    }
    // block reduce sum-of-squares
    red[tid] = loc;
    __syncthreads();
    for (int s = 128; s > 0; s >>= 1) {
      if (tid < s) red[tid] += red[tid + s];
      __syncthreads();
    }
    float n = red[0];
    float scale = n / ((1.0f + n) * sqrtf(1e-9f + n));
    for (int j = 0; j < 3; ++j) cl[j] *= scale; // squash in place

    // b[k] += sum_d eh[k][d] * c_hat[d]
    float bp[KK];
    for (int k = 0; k < KK; ++k) {
      float s = 0.f;
      for (int j = 0; j < 3; ++j) s += eh[k * DD + tid + j * 256] * cl[j];
      bp[k] = s;
    }
    if (tid < KK) bacc[tid] = 0.0f;
    __syncthreads();
    for (int k = 0; k < KK; ++k) atomicAdd(&bacc[k], bp[k]);
    __syncthreads();
    if (tid < KK) blog[tid] += bacc[tid];
    __syncthreads();
  }

  // final c_i (last iteration's squash) -> bf16
  for (int j = 0; j < 3; ++j)
    cib[(size_t)blockIdx.x * DD + tid + j * 256] = f2bf(cl[j]);
}

// ---------------------------------------------------------------------------
// Kernel 3: qs = qf . c_i^T per batch, fused sigmoid(qs * sum(fc_w) + fc_b).
// One wave (32 threads) per (batch, 16-row tile); 2 WMMA accumulators cover
// class columns 0..15 and 16..31 (only 0..19 stored).  Query (236 MB) is the
// dominant HBM stream; it is read exactly once, converted in-register.
// ---------------------------------------------------------------------------
__global__ __launch_bounds__(32) void k_gemm2(const float* __restrict__ Qm,
                                              const unsigned short* __restrict__ cib,
                                              const float* __restrict__ fcw,
                                              const float* __restrict__ fcb,
                                              float* __restrict__ out) {
  const int b = blockIdx.x / RTILES;
  const int t = blockIdx.x - b * RTILES;
  const int l = threadIdx.x, lm = l & 15, lh = l >> 4;

  float sfw = 0.f;
  for (int i = 0; i < HH; ++i) sfw += fcw[i]; // uniform -> scalar loads
  const float fb = fcb[0];

  int rrow = t * 16 + lm;                             // 0..303 within batch
  int grow = b * QPB + (rrow < QPB ? rrow : QPB - 1); // clamp (result discarded)
  const float* arow = Qm + (size_t)grow * DD;

  int c0 = lm, c1 = 16 + lm;
  const unsigned short* br0 = cib + (size_t)(b * CC + (c0 < CC ? c0 : CC - 1)) * DD;
  const unsigned short* br1 = cib + (size_t)(b * CC + (c1 < CC ? c1 : CC - 1)) * DD;

  const v8f zero = {0.f, 0.f, 0.f, 0.f, 0.f, 0.f, 0.f, 0.f};
  v8f acc0 = zero, acc1 = zero;

  for (int kk = 0; kk < DD; kk += 32) {
    const float* ap = arow + kk + lh * 16;
    float4 f0 = *(const float4*)(ap);
    float4 f1 = *(const float4*)(ap + 4);
    float4 f2 = *(const float4*)(ap + 8);
    float4 f3 = *(const float4*)(ap + 12);
    v16us au;
    au[0] = f2bf(f0.x);  au[1] = f2bf(f0.y);  au[2] = f2bf(f0.z);  au[3] = f2bf(f0.w);
    au[4] = f2bf(f1.x);  au[5] = f2bf(f1.y);  au[6] = f2bf(f1.z);  au[7] = f2bf(f1.w);
    au[8] = f2bf(f2.x);  au[9] = f2bf(f2.y);  au[10] = f2bf(f2.z); au[11] = f2bf(f2.w);
    au[12] = f2bf(f3.x); au[13] = f2bf(f3.y); au[14] = f2bf(f3.z); au[15] = f2bf(f3.w);
    v16bf a = __builtin_bit_cast(v16bf, au);

    v16bf b0 = __builtin_bit_cast(v16bf, *(const v16us*)(br0 + kk + lh * 16));
    v16bf b1 = __builtin_bit_cast(v16bf, *(const v16us*)(br1 + kk + lh * 16));

    acc0 = __builtin_amdgcn_wmma_f32_16x16x32_bf16(false, a, false, b0,
                                                   (short)0, acc0, false, false);
    acc1 = __builtin_amdgcn_wmma_f32_16x16x32_bf16(false, a, false, b1,
                                                   (short)0, acc1, false, false);
  }

  for (int v = 0; v < 8; ++v) {
    int r = t * 16 + lh * 8 + v;
    if (r < QPB) {
      size_t orow = ((size_t)b * QPB + r) * CC;
      float s0 = 1.0f / (1.0f + __expf(-(acc0[v] * sfw + fb)));
      out[orow + c0] = s0;            // c0 = 0..15 < 20 always valid
      if (c1 < CC) {
        float s1 = 1.0f / (1.0f + __expf(-(acc1[v] * sfw + fb)));
        out[orow + c1] = s1;
      }
    }
  }
}

// ---------------------------------------------------------------------------
// Kernel 4: first-max argmax over the 20 classes per row.
// ---------------------------------------------------------------------------
__global__ void k_argmax(const float* __restrict__ score, int* __restrict__ pred,
                         int nrows) {
  int r = blockIdx.x * 256 + threadIdx.x;
  if (r >= nrows) return;
  const float* row = score + (size_t)r * CC;
  float best = row[0];
  int bi = 0;
  for (int c = 1; c < CC; ++c) {
    float v = row[c];
    if (v > best) { best = v; bi = c; } // strict > keeps first max (jnp.argmax)
  }
  pred[r] = bi;
}

// ---------------------------------------------------------------------------
// Workspace layout: [ WT bf16 | e_hat f32 | c_i bf16 ]  (~88 MB total)
// ---------------------------------------------------------------------------
#define WT_BYTES ((size_t)DD * DD * 2)              //  1,179,648
#define EH_BYTES ((size_t)SROWS * DD * 4)           // 78,643,200
#define CI_BYTES ((size_t)BB * CC * DD * 2)         //  7,864,320

extern "C" void kernel_launch(void* const* d_in, const int* in_sizes, int n_in,
                              void* d_out, int out_size, void* d_ws, size_t ws_size,
                              hipStream_t stream) {
  const float* support = (const float*)d_in[0];
  const float* query   = (const float*)d_in[1];
  const float* W_w     = (const float*)d_in[2];
  const float* W_b     = (const float*)d_in[3];
  // d_in[4] = M : unused by the reference computation
  const float* fc_w    = (const float*)d_in[5];
  const float* fc_b    = (const float*)d_in[6];
  (void)in_sizes; (void)n_in; (void)out_size; (void)ws_size;

  unsigned short* WT   = (unsigned short*)d_ws;
  float*          ehat = (float*)((char*)d_ws + WT_BYTES);
  unsigned short* cib  = (unsigned short*)((char*)d_ws + WT_BYTES + EH_BYTES);

  float* score = (float*)d_out;
  int*   pred  = (int*)(score + (size_t)QROWS * CC);

  k_wt    <<<(DD * DD + 255) / 256, 256, 0, stream>>>(W_w, WT);
  k_gemm1 <<<SROWS / MROWS,         512, 0, stream>>>(support, WT, W_b, ehat);
  k_route <<<BB * CC,               256, 0, stream>>>(ehat, cib);
  k_gemm2 <<<BB * RTILES,            32, 0, stream>>>(query, cib, fc_w, fc_b, score);
  k_argmax<<<(QROWS + 255) / 256,   256, 0, stream>>>(score, pred, QROWS);
}
